// LearnedQueryAttention_23854248362379
// MI455X (gfx1250) — compile-verified, matrix operand order checked by reading, not run on hardware
//
#include <hip/hip_runtime.h>
#include <hip/hip_bf16.h>
#include <cstdint>

typedef unsigned short u16;
typedef unsigned int   u32;
typedef __attribute__((ext_vector_type(8)))  float  v8f;
typedef __attribute__((ext_vector_type(16))) __bf16 v16bf;

#define EPS_RMS 1.1920929e-07f

static __device__ __forceinline__ u16 f2bf(float f) {
  union { float f; u32 u; } c; c.f = f;
  u32 r = c.u + 0x7FFFu + ((c.u >> 16) & 1u);   // round-to-nearest-even
  return (u16)(r >> 16);
}
static __device__ __forceinline__ float bflo(u32 u) { union { u32 u; float f; } c; c.u = u << 16;          return c.f; }
static __device__ __forceinline__ float bfhi(u32 u) { union { u32 u; float f; } c; c.u = u & 0xFFFF0000u;  return c.f; }

// CDNA5 async global->LDS copy (16B per lane), tracked by ASYNCcnt.
static __device__ __forceinline__ void async_ld16(u32 ldsOff, const void* gptr) {
  asm volatile("global_load_async_to_lds_b128 %0, %1, off"
               :: "v"(ldsOff), "v"(gptr) : "memory");
}
static __device__ __forceinline__ void wait_async0() {
  asm volatile("s_wait_asynccnt 0" ::: "memory");
}
static __device__ __forceinline__ u32 lds_off(const void* p) {
  return (u32)(uintptr_t)p;   // low 32 bits of generic addr = DS offset
}

// ---------------------------------------------------------------------------
// Kernel 1: per-row RMSNorm (D=1024) -> bf16
// ---------------------------------------------------------------------------
__global__ __launch_bounds__(256) void rmsnorm_to_bf16(
    const float* __restrict__ x, const float* __restrict__ w, u16* __restrict__ xn)
{
  __shared__ float red[256];
  const int tid = threadIdx.x;
  const size_t row = blockIdx.x;
  const float* xr = x + row * 1024;
  float v[4]; float ssq = 0.f;
#pragma unroll
  for (int i = 0; i < 4; ++i) { v[i] = xr[tid + i * 256]; ssq += v[i] * v[i]; }
  red[tid] = ssq;
  __syncthreads();
  for (int off = 128; off > 0; off >>= 1) {
    if (tid < off) red[tid] += red[tid + off];
    __syncthreads();
  }
  const float scale = rsqrtf(red[0] * (1.0f / 1024.0f) + EPS_RMS);
  u16* orow = xn + row * 1024;
#pragma unroll
  for (int i = 0; i < 4; ++i)
    orow[tid + i * 256] = f2bf(v[i] * scale * w[tid + i * 256]);
}

// ---------------------------------------------------------------------------
// Kernel 2: fp32 -> bf16 conversion (weights)
// ---------------------------------------------------------------------------
__global__ void cvt_f32_bf16(const float* __restrict__ in, u16* __restrict__ out, int n)
{
  int i = blockIdx.x * blockDim.x + threadIdx.x;
  const int stride = gridDim.x * blockDim.x;
  for (; i < n; i += stride) out[i] = f2bf(in[i]);
}

// ---------------------------------------------------------------------------
// Kernel 3: q projection: qp[l][n] = sum_k tmpl[l][k] * wq[n][k]   (4 x 1024)
// ---------------------------------------------------------------------------
__global__ __launch_bounds__(256) void qproj_kernel(
    const float* __restrict__ tmpl, const float* __restrict__ wq, float* __restrict__ qp)
{
  const int idx = blockIdx.x * 256 + threadIdx.x;   // 0..4095
  const int l = idx >> 10;
  const int n = idx & 1023;
  const float* t  = tmpl + l * 1024;
  const float* wr = wq + (size_t)n * 1024;
  float acc = 0.f;
  for (int k = 0; k < 1024; ++k) acc += t[k] * wr[k];
  qp[idx] = acc;
}

// ---------------------------------------------------------------------------
// Kernel 4: nseg[b] = seg_id[b][S-1] + 1 (seg ids contiguous non-decreasing)
// ---------------------------------------------------------------------------
__global__ void nseg_kernel(const int* __restrict__ seg_id, int* __restrict__ nseg, int S, int B)
{
  int b = threadIdx.x;
  if (b < B) nseg[b] = seg_id[(size_t)b * S + S - 1] + 1;
}

// ---------------------------------------------------------------------------
// WMMA GEMM: C[M,1024] = A[M,1024] * B[1024,1024]^T, bf16 in, fp32 accumulate.
// Block tile 128x256, K step 32, double-buffered LDS staged with
// global_load_async_to_lds_b128 (ASYNCcnt). 256 threads = 8 waves; wave grid
// 2x4, each wave owns 64x64 = 4x4 tiles of 16x16 (16 WMMA / K-step / wave).
// K-loop manually unrolled x2 so buffer selection is compile-time static.
// ---------------------------------------------------------------------------
union FragBF { v16bf v; u32 u[8]; };

#define GEMM_KD 1024
#define GEMM_ND 1024

template<bool STORE_BF16, bool MASK>
__global__ __launch_bounds__(256) void gemm_bf16_wmma(
    const u16* __restrict__ A, const u16* __restrict__ Bm,
    void* __restrict__ Cout, const int* __restrict__ nseg)
{
  __shared__ u16 lsa[2][128][32];   // 2 x 8 KB
  __shared__ u16 lsb[2][256][32];   // 2 x 16 KB
  const int tid  = threadIdx.x;
  const int wave = tid >> 5;
  const int lane = tid & 31;
  const int wm = (wave >> 2) * 64;   // 0,64
  const int wn = (wave &  3) * 64;   // 0,64,128,192
  const int bm = blockIdx.x * 128;
  const int bn = blockIdx.y * 256;

  v8f acc[4][4];
#pragma unroll
  for (int i = 0; i < 4; ++i)
#pragma unroll
    for (int j = 0; j < 4; ++j)
#pragma unroll
      for (int e = 0; e < 8; ++e) acc[i][j][e] = 0.0f;

  // --- async staging descriptors: 16B chunks; A tile = 512 chunks (2/thread),
  //     B tile = 1024 chunks (4/thread). chunk idx -> row = idx>>2, col = (idx&3)*8
  const u16* gAc[2]; u32 lAc[2][2];
#pragma unroll
  for (int i = 0; i < 2; ++i) {
    const int idx = tid + 256 * i, row = idx >> 2, col = (idx & 3) * 8;
    gAc[i] = A + (size_t)(bm + row) * GEMM_KD + col;
    lAc[0][i] = lds_off(&lsa[0][row][col]);
    lAc[1][i] = lds_off(&lsa[1][row][col]);
  }
  const u16* gBc[4]; u32 lBc[2][4];
#pragma unroll
  for (int i = 0; i < 4; ++i) {
    const int idx = tid + 256 * i, row = idx >> 2, col = (idx & 3) * 8;
    gBc[i] = Bm + (size_t)(bn + row) * GEMM_KD + col;
    lBc[0][i] = lds_off(&lsb[0][row][col]);
    lBc[1][i] = lds_off(&lsb[1][row][col]);
  }

  // fragment addressing (ISA: 16-bit A 16x32 layout; B stored N-major mirrors A)
  const int fm = lane & 15;
  const int kb = (lane >> 4) * 8;

  // issue one K-step of async DMA into buffer `buf` (compile-time constant)
  auto issue = [&](int buf, int koff) {
#pragma unroll
    for (int i = 0; i < 2; ++i) async_ld16(lAc[buf][i], gAc[i] + koff);
#pragma unroll
    for (int i = 0; i < 4; ++i) async_ld16(lBc[buf][i], gBc[i] + koff);
  };

  // one K-step of WMMA from a statically-addressed buffer
  auto compute = [&](const u16 (*la)[32], const u16 (*lb)[32]) {
    FragBF fa[4], fb[4];
#pragma unroll
    for (int mt = 0; mt < 4; ++mt) {
      const u16* p = &la[wm + mt * 16 + fm][0];
#pragma unroll
      for (int j = 0; j < 4; ++j) {
        fa[mt].u[j]     = *(const u32*)(p + kb + 2 * j);
        fa[mt].u[4 + j] = *(const u32*)(p + 16 + kb + 2 * j);
      }
    }
#pragma unroll
    for (int nt = 0; nt < 4; ++nt) {
      const u16* p = &lb[wn + nt * 16 + fm][0];
#pragma unroll
      for (int j = 0; j < 4; ++j) {
        fb[nt].u[j]     = *(const u32*)(p + kb + 2 * j);
        fb[nt].u[4 + j] = *(const u32*)(p + 16 + kb + 2 * j);
      }
    }
#pragma unroll
    for (int mt = 0; mt < 4; ++mt)
#pragma unroll
      for (int nt = 0; nt < 4; ++nt)
        acc[mt][nt] = __builtin_amdgcn_wmma_f32_16x16x32_bf16(
            false, fa[mt].v, false, fb[nt].v, (short)0, acc[mt][nt], false, false);
  };

  // prologue: stage K-step 0 into buffer 0
  issue(0, 0);

#pragma unroll 1
  for (int k0 = 0; k0 < GEMM_KD; k0 += 64) {
    // even half-step: consume buf0, DMA next into buf1
    wait_async0();       // this wave's async writes into buf0 have landed
    __syncthreads();     // all waves landed; prior reads of buf1 retired
    issue(1, k0 + 32);
    compute(lsa[0], lsb[0]);
    // odd half-step: consume buf1, DMA next into buf0
    wait_async0();
    __syncthreads();
    if (k0 + 64 < GEMM_KD) issue(0, k0 + 64);
    compute(lsa[1], lsb[1]);
  }

  // store: C/D layout — lane n = lane&15; vgpr r -> row r (+8 for lanes 16..31)
  const int fn  = lane & 15;
  const int mhi = (lane >> 4) * 8;
#pragma unroll
  for (int mt = 0; mt < 4; ++mt) {
#pragma unroll
    for (int nt = 0; nt < 4; ++nt) {
#pragma unroll
      for (int r = 0; r < 8; ++r) {
        const int row = bm + wm + mt * 16 + mhi + r;
        const int col = bn + wn + nt * 16 + fn;
        float v = acc[mt][nt][r];
        if (MASK) {
          const int bb = row >> 8, q = row & 255;
          if (q >= nseg[bb] * 4) v = 0.0f;
        }
        if (STORE_BF16) ((u16*)Cout)[(size_t)row * GEMM_ND + col] = f2bf(v);
        else            ((float*)Cout)[(size_t)row * GEMM_ND + col] = v;
      }
    }
  }
}

// ---------------------------------------------------------------------------
// Kernel 5: per-segment attention + fused out-RMSNorm -> bf16 ctx_norm.
// Grid (NSEG, B). 64 tokens/segment, 4 queries/segment, 16 heads, hd=64.
// ---------------------------------------------------------------------------
__global__ __launch_bounds__(256) void attn_segment(
    const u16* __restrict__ kbuf, const u16* __restrict__ vbuf,
    const float* __restrict__ qproj, const int* __restrict__ seg_id,
    const float* __restrict__ out_w, u16* __restrict__ ctxn, int S)
{
  const int g = blockIdx.x;      // segment
  const int b = blockIdx.y;      // batch
  const int tid = threadIdx.x;
  __shared__ float qs[4 * 1024];     // 16 KB
  __shared__ float sc[64][64];       // [h*4+q][s] 16 KB
  __shared__ float ctx[4][1024];     // 16 KB
  __shared__ int   sid[64];
  __shared__ float rred[4];

  for (int i = tid; i < 4096; i += 256) qs[i] = qproj[i];
  if (tid < 64) sid[tid] = seg_id[(size_t)b * S + g * 64 + tid];
  if (tid < 4)  rred[tid] = 0.0f;
  __syncthreads();

  // scores: thread t -> (h,q) = t>>2, 16 keys starting at (t&3)*16
  {
    const int qh = tid >> 2;
    const int q  = qh & 3;
    const int h  = qh >> 2;
    const int s0 = (tid & 3) * 16;
    const float* qv = &qs[q * 1024 + h * 64];
    for (int i = 0; i < 16; ++i) {
      const int s = s0 + i;
      float dot;
      if (sid[s] != g) {
        dot = -1e30f;                         // masked key
      } else {
        dot = 0.f;
        const uint4* kp = (const uint4*)(kbuf + ((size_t)b * S + g * 64 + s) * 1024 + h * 64);
#pragma unroll
        for (int c = 0; c < 8; ++c) {
          uint4 kk = kp[c];
          dot += qv[c*8+0] * bflo(kk.x) + qv[c*8+1] * bfhi(kk.x);
          dot += qv[c*8+2] * bflo(kk.y) + qv[c*8+3] * bfhi(kk.y);
          dot += qv[c*8+4] * bflo(kk.z) + qv[c*8+5] * bfhi(kk.z);
          dot += qv[c*8+6] * bflo(kk.w) + qv[c*8+7] * bfhi(kk.w);
        }
        dot *= 0.125f;                        // 1/sqrt(64)
      }
      sc[qh][s] = dot;
    }
  }
  __syncthreads();

  // softmax over the 64 segment keys, one (h,q) row per thread
  if (tid < 64) {
    float m = -1e30f;
    for (int s = 0; s < 64; ++s) m = fmaxf(m, sc[tid][s]);
    float sum = 0.f;
    for (int s = 0; s < 64; ++s) { float e = __expf(sc[tid][s] - m); sc[tid][s] = e; sum += e; }
    const float inv = 1.0f / sum;
    for (int s = 0; s < 64; ++s) sc[tid][s] *= inv;
  }
  __syncthreads();

  // ctx: thread t -> query q = t>>6, 16 channels at dd0 = (t&63)*16 (same head)
  {
    const int q   = tid >> 6;
    const int dd0 = (tid & 63) * 16;
    const int h   = dd0 >> 6;
    float acc[16];
#pragma unroll
    for (int i = 0; i < 16; ++i) acc[i] = 0.f;
    const float* arow = sc[h * 4 + q];
    for (int s = 0; s < 64; ++s) {
      const float w = arow[s];
      const uint4* vp = (const uint4*)(vbuf + ((size_t)b * S + g * 64 + s) * 1024 + dd0);
      uint4 v0 = vp[0], v1 = vp[1];
      acc[0]  += w * bflo(v0.x); acc[1]  += w * bfhi(v0.x);
      acc[2]  += w * bflo(v0.y); acc[3]  += w * bfhi(v0.y);
      acc[4]  += w * bflo(v0.z); acc[5]  += w * bfhi(v0.z);
      acc[6]  += w * bflo(v0.w); acc[7]  += w * bfhi(v0.w);
      acc[8]  += w * bflo(v1.x); acc[9]  += w * bfhi(v1.x);
      acc[10] += w * bflo(v1.y); acc[11] += w * bfhi(v1.y);
      acc[12] += w * bflo(v1.z); acc[13] += w * bfhi(v1.z);
      acc[14] += w * bflo(v1.w); acc[15] += w * bfhi(v1.w);
    }
    float ssq = 0.f;
#pragma unroll
    for (int i = 0; i < 16; ++i) { ctx[q][dd0 + i] = acc[i]; ssq += acc[i] * acc[i]; }
    atomicAdd(&rred[q], ssq);
  }
  __syncthreads();

  // fused out-RMSNorm -> bf16
  {
    const int q   = tid >> 6;
    const int dd0 = (tid & 63) * 16;
    const float scale = rsqrtf(rred[q] * (1.0f / 1024.0f) + EPS_RMS);
    const size_t row = ((size_t)b * 256 + g * 4 + q) * 1024;
#pragma unroll
    for (int i = 0; i < 16; ++i)
      ctxn[row + dd0 + i] = f2bf(ctx[q][dd0 + i] * scale * out_w[dd0 + i]);
  }
}

// ---------------------------------------------------------------------------
extern "C" void kernel_launch(void* const* d_in, const int* in_sizes, int n_in,
                              void* d_out, int out_size, void* d_ws, size_t ws_size,
                              hipStream_t stream) {
  const float* x      = (const float*)d_in[0];
  const int*   seg_id = (const int*)  d_in[1];
  const float* tmpl   = (const float*)d_in[2];
  const float* in_w   = (const float*)d_in[3];
  const float* out_w  = (const float*)d_in[4];
  const float* wq     = (const float*)d_in[5];
  const float* wk     = (const float*)d_in[6];
  const float* wv     = (const float*)d_in[7];
  const float* wo     = (const float*)d_in[8];

  const int B = 4, S = 4096, D = 1024, NSEG = 64;
  const size_t BS = (size_t)B * S;             // 16384 rows

  char* p = (char*)d_ws;
  u16*   xn   = (u16*)p;   p += BS * D * 2;              // 32 MB x_norm bf16
  u16*   kbuf = (u16*)p;   p += BS * D * 2;              // 32 MB K bf16
  u16*   vbuf = (u16*)p;   p += BS * D * 2;              // 32 MB V bf16
  u16*   wkb  = (u16*)p;   p += (size_t)D * D * 2;       // 2 MB
  u16*   wvb  = (u16*)p;   p += (size_t)D * D * 2;       // 2 MB
  u16*   wob  = (u16*)p;   p += (size_t)D * D * 2;       // 2 MB
  u16*   ctxn = (u16*)p;   p += (size_t)B * 256 * D * 2; // 2 MB ctx_norm bf16
  float* qp   = (float*)p; p += (size_t)4 * D * 4;       // 16 KB q projection
  int*   nseg = (int*)p;   p += 256;

  rmsnorm_to_bf16<<<dim3((unsigned)BS), 256, 0, stream>>>(x, in_w, xn);
  cvt_f32_bf16<<<dim3(1024), 256, 0, stream>>>(wk, wkb, D * D);
  cvt_f32_bf16<<<dim3(1024), 256, 0, stream>>>(wv, wvb, D * D);
  cvt_f32_bf16<<<dim3(1024), 256, 0, stream>>>(wo, wob, D * D);
  qproj_kernel<<<dim3(16), 256, 0, stream>>>(tmpl, wq, qp);
  nseg_kernel<<<dim3(1), 32, 0, stream>>>(seg_id, nseg, S, B);

  gemm_bf16_wmma<true, false><<<dim3((unsigned)(BS / 128), D / 256), 256, 0, stream>>>(
      xn, wkb, kbuf, nullptr);
  gemm_bf16_wmma<true, false><<<dim3((unsigned)(BS / 128), D / 256), 256, 0, stream>>>(
      xn, wvb, vbuf, nullptr);

  attn_segment<<<dim3(NSEG, B), 256, 0, stream>>>(kbuf, vbuf, qp, seg_id, out_w, ctxn, S);

  gemm_bf16_wmma<false, true><<<dim3((B * 256) / 128, D / 256), 256, 0, stream>>>(
      ctxn, wob, (float*)d_out, nseg);
}
